// MultiHeadSelfAttention_22771916603463
// MI455X (gfx1250) — compile-verified, matrix operand order checked by reading, not run on hardware
//
#include <hip/hip_runtime.h>
#include <hip/hip_bf16.h>

#define DM   1024
#define NH   16
#define DKH  64
#define SEQ  2048
#define BATCH 2
#define MROWS (BATCH*SEQ)   // 4096

typedef unsigned short u16;
typedef unsigned int   u32;
typedef __attribute__((ext_vector_type(16))) __bf16 v16bf;
typedef __attribute__((ext_vector_type(8)))  float  v8f;
typedef __attribute__((ext_vector_type(8)))  u16    u16x8;

union FragU { u16x8 h[2]; v16bf f; };

__device__ __forceinline__ u16 f2bf(float x) {
  union { float f; u32 u; } c; c.f = x;
  u32 r = c.u + 0x7FFFu + ((c.u >> 16) & 1u);   // round-to-nearest-even
  return (u16)(r >> 16);
}

// Per-lane fragment load for 16x32 bf16 A/B fragments (wave32 layout):
// element e -> K = (e%8) + 16*(e/8) + 8*half  => two contiguous 16B chunks.
__device__ __forceinline__ v16bf load_frag(const u16* base, int hlf) {
  FragU u;
  u.h[0] = *(const u16x8*)(base + 8*hlf);
  u.h[1] = *(const u16x8*)(base + 16 + 8*hlf);
  return u.f;
}

#define WMMA_BF16(a,b,c) \
  __builtin_amdgcn_wmma_f32_16x16x32_bf16(false,(a),false,(b),(short)0,(c),false,false)

__global__ __launch_bounds__(256)
void convert_f32_to_bf16(const float* __restrict__ src, u16* __restrict__ dst, int n) {
  int i = blockIdx.x*blockDim.x + threadIdx.x;
  int stride = gridDim.x*blockDim.x;
  for (; i < n; i += stride) dst[i] = f2bf(src[i]);
}

// Y = A (Mx1024 bf16, row-major) * W^T (W is 1024x1024 bf16 row-major) + bias.
// MODE 0: bf16 row-major out. MODE 1: per-head transposed Vt[b,h,d,s] out.
// MODE 2: fp32 row-major out. Each wave computes a 64x64 tile (16 WMMA/chunk).
template<int MODE>
__global__ __launch_bounds__(128)
void gemm_xwt(const u16* __restrict__ A, const u16* __restrict__ W,
              const float* __restrict__ bias, void* __restrict__ out) {
  const int lane = threadIdx.x & 31;
  const int hlf  = lane >> 4;
  const int l16  = lane & 15;
  const int wid  = blockIdx.x * 4 + (threadIdx.x >> 5);   // 1024 wave tasks
  const int r0   = (wid >> 4) * 64;                       // 64 row tiles of 64
  const int c0   = (wid & 15) * 64;                       // 16 col tiles of 64

  v8f acc[4][4] = {};
  const u16* arow[4];
  const u16* wrow[4];
  #pragma unroll
  for (int i = 0; i < 4; ++i) arow[i] = A + (size_t)(r0 + i*16 + l16) * DM;
  #pragma unroll
  for (int j = 0; j < 4; ++j) wrow[j] = W + (size_t)(c0 + j*16 + l16) * DM;

  for (int k0 = 0; k0 < DM; k0 += 32) {
    __builtin_prefetch(arow[0] + k0 + 64, 0, 0);
    __builtin_prefetch(arow[2] + k0 + 64, 0, 0);
    v16bf a[4];
    #pragma unroll
    for (int i = 0; i < 4; ++i) a[i] = load_frag(arow[i] + k0, hlf);
    #pragma unroll
    for (int j = 0; j < 4; ++j) {
      v16bf bf = load_frag(wrow[j] + k0, hlf);
      #pragma unroll
      for (int i = 0; i < 4; ++i) acc[i][j] = WMMA_BF16(a[i], bf, acc[i][j]);
    }
  }

  #pragma unroll
  for (int i = 0; i < 4; ++i) {
    #pragma unroll
    for (int j = 0; j < 4; ++j) {
      const int col = c0 + j*16 + l16;
      const float bb = bias[col];
      if (MODE == 1) {
        // C-frag rows (r=0..7) are consecutive tokens s -> pack one b128 store
        const int nbase = r0 + i*16 + 8*hlf;
        const int b = nbase >> 11, s0 = nbase & (SEQ - 1);
        const int h = col >> 6,   d  = col & (DKH - 1);
        u16x8 pack;
        #pragma unroll
        for (int r = 0; r < 8; ++r) pack[r] = f2bf(acc[i][j][r] + bb);
        *(u16x8*)((u16*)out + (((size_t)(b*NH + h) * DKH + d) * SEQ + s0)) = pack;
      } else {
        #pragma unroll
        for (int r = 0; r < 8; ++r) {
          const int row = r0 + i*16 + 8*hlf + r;
          const float v = acc[i][j][r] + bb;
          if (MODE == 0) ((u16*)out)[(size_t)row * DM + col] = f2bf(v);
          else           ((float*)out)[(size_t)row * DM + col] = v;
        }
      }
    }
  }
}

// Flash-style attention: one wave = 32 query rows of one (b,h), 64 keys/iter.
// Q,K: bf16 row-major (4096x1024). Vt: bf16 [b,h,d,s]. Ctx: bf16 row-major.
// K fragments are consumed immediately after load (short live ranges) to stay
// below 256 VGPRs and avoid s_set_vgpr_msb churn.
__global__ __launch_bounds__(128)
void attn_flash(const u16* __restrict__ Q, const u16* __restrict__ Kc,
                const u16* __restrict__ Vt, const int* __restrict__ mask,
                u16* __restrict__ Ctx) {
  __shared__ u16 plds[4][2][16 * 64];   // per-wave: 2 q-halves x 16x64 P tile
  const int lane = threadIdx.x & 31;
  const int hlf  = lane >> 4;
  const int l16  = lane & 15;
  const int w    = threadIdx.x >> 5;
  const int wid  = blockIdx.x * 4 + w;          // 2048 wave tasks
  const int qt = wid & 63;
  const int h  = (wid >> 6) & 15;
  const int b  = wid >> 10;
  const int q0 = qt * 32;

  v16bf qf[2][2];
  #pragma unroll
  for (int i = 0; i < 2; ++i)
    #pragma unroll
    for (int t = 0; t < 2; ++t)
      qf[i][t] = load_frag(Q + (size_t)(b*SEQ + q0 + i*16 + l16) * DM + h*DKH + t*32, hlf);

  float mrow[2][8], lrow[2][8];
  #pragma unroll
  for (int i = 0; i < 2; ++i)
    #pragma unroll
    for (int r = 0; r < 8; ++r) { mrow[i][r] = -1e30f; lrow[i][r] = 0.0f; }
  v8f o[2][4] = {};

  for (int kb = 0; kb < SEQ; kb += 64) {
    // ---- scores: per key tile j, load K frags then consume for both halves ----
    v8f s[2][4];
    #pragma unroll
    for (int j = 0; j < 4; ++j) {
      const u16* krow = Kc + (size_t)(b*SEQ + kb + j*16 + l16) * DM + h*DKH;
      v16bf kf0 = load_frag(krow, hlf);
      v16bf kf1 = load_frag(krow + 32, hlf);
      #pragma unroll
      for (int i = 0; i < 2; ++i) {
        v8f z = {};
        z = WMMA_BF16(qf[i][0], kf0, z);
        s[i][j] = WMMA_BF16(qf[i][1], kf1, z);
      }
    }

    int mk[4];
    #pragma unroll
    for (int j = 0; j < 4; ++j) mk[j] = mask[b*SEQ + kb + j*16 + l16];

    // ---- online softmax per q-half ----
    #pragma unroll
    for (int i = 0; i < 2; ++i) {
      float rm[8];
      #pragma unroll
      for (int r = 0; r < 8; ++r) {
        float v0 = mk[0] ? s[i][0][r] * 0.125f : -1e9f;   // 1/sqrt(64)
        float v1 = mk[1] ? s[i][1][r] * 0.125f : -1e9f;
        float v2 = mk[2] ? s[i][2][r] * 0.125f : -1e9f;
        float v3 = mk[3] ? s[i][3][r] * 0.125f : -1e9f;
        s[i][0][r] = v0; s[i][1][r] = v1; s[i][2][r] = v2; s[i][3][r] = v3;
        rm[r] = fmaxf(fmaxf(v0, v1), fmaxf(v2, v3));
      }
      #pragma unroll
      for (int r = 0; r < 8; ++r)
        #pragma unroll
        for (int t2 = 1; t2 < 16; t2 <<= 1)
          rm[r] = fmaxf(rm[r], __shfl_xor(rm[r], t2, 32));

      u16* myp = &plds[w][i][0];
      #pragma unroll
      for (int r = 0; r < 8; ++r) {
        const float mn    = fmaxf(mrow[i][r], rm[r]);
        const float alpha = __expf(mrow[i][r] - mn);
        mrow[i][r] = mn;
        float rs = 0.0f;
        #pragma unroll
        for (int j = 0; j < 4; ++j) {
          const float p = __expf(s[i][j][r] - mn);
          myp[(r + 8*hlf)*64 + j*16 + l16] = f2bf(p);   // C-frag -> A-frag via LDS
          rs += p;
        }
        #pragma unroll
        for (int t2 = 1; t2 < 16; t2 <<= 1) rs += __shfl_xor(rs, t2, 32);
        lrow[i][r] = lrow[i][r]*alpha + rs;
        #pragma unroll
        for (int n = 0; n < 4; ++n) o[i][n][r] *= alpha;
      }
    }

    // ---- P (2 x 16x64) @ V (64x64): V frags shared across both q-halves ----
    #pragma unroll
    for (int t = 0; t < 2; ++t) {
      v16bf pf0 = load_frag(&plds[w][0][l16*64 + t*32], hlf);
      v16bf pf1 = load_frag(&plds[w][1][l16*64 + t*32], hlf);
      #pragma unroll
      for (int n = 0; n < 4; ++n) {
        v16bf vf = load_frag(Vt + (size_t)((b*NH + h)*DKH + n*16 + l16) * SEQ + kb + t*32, hlf);
        o[0][n] = WMMA_BF16(pf0, vf, o[0][n]);
        o[1][n] = WMMA_BF16(pf1, vf, o[1][n]);
      }
    }
  }

  #pragma unroll
  for (int i = 0; i < 2; ++i) {
    float inv[8];
    #pragma unroll
    for (int r = 0; r < 8; ++r) inv[r] = 1.0f / fmaxf(lrow[i][r], 1e-30f);
    #pragma unroll
    for (int n = 0; n < 4; ++n)
      #pragma unroll
      for (int r = 0; r < 8; ++r) {
        const int row = b*SEQ + q0 + i*16 + r + 8*hlf;
        const int col = h*DKH + n*16 + l16;
        Ctx[(size_t)row * DM + col] = f2bf(o[i][n][r] * inv[r]);
      }
  }
}

extern "C" void kernel_launch(void* const* d_in, const int* in_sizes, int n_in,
                              void* d_out, int out_size, void* d_ws, size_t ws_size,
                              hipStream_t stream) {
  (void)in_sizes; (void)n_in; (void)out_size; (void)ws_size;
  const float* x    = (const float*)d_in[0];
  const int*   mask = (const int*)d_in[1];
  const float* Wq = (const float*)d_in[2];
  const float* bq = (const float*)d_in[3];
  const float* Wk = (const float*)d_in[4];
  const float* bk = (const float*)d_in[5];
  const float* Wv = (const float*)d_in[6];
  const float* bv = (const float*)d_in[7];
  const float* Wo = (const float*)d_in[8];
  const float* bo = (const float*)d_in[9];
  float* out = (float*)d_out;

  u16* xb  = (u16*)d_ws;
  u16* Wqb = xb  + (size_t)MROWS*DM;
  u16* Wkb = Wqb + (size_t)DM*DM;
  u16* Wvb = Wkb + (size_t)DM*DM;
  u16* Wob = Wvb + (size_t)DM*DM;
  u16* Qb  = Wob + (size_t)DM*DM;
  u16* Kb  = Qb  + (size_t)MROWS*DM;
  u16* Vt  = Kb  + (size_t)MROWS*DM;
  u16* Ctx = Vt  + (size_t)MROWS*DM;

  convert_f32_to_bf16<<<1024, 256, 0, stream>>>(x,  xb,  MROWS*DM);
  convert_f32_to_bf16<<<256,  256, 0, stream>>>(Wq, Wqb, DM*DM);
  convert_f32_to_bf16<<<256,  256, 0, stream>>>(Wk, Wkb, DM*DM);
  convert_f32_to_bf16<<<256,  256, 0, stream>>>(Wv, Wvb, DM*DM);
  convert_f32_to_bf16<<<256,  256, 0, stream>>>(Wo, Wob, DM*DM);

  gemm_xwt<0><<<256, 128, 0, stream>>>(xb, Wqb, bq, Qb);
  gemm_xwt<0><<<256, 128, 0, stream>>>(xb, Wkb, bk, Kb);
  gemm_xwt<1><<<256, 128, 0, stream>>>(xb, Wvb, bv, Vt);

  attn_flash<<<512, 128, 0, stream>>>(Qb, Kb, Vt, mask, Ctx);

  gemm_xwt<2><<<256, 128, 0, stream>>>(Ctx, Wob, bo, out);
}
